// ConvAttention_54855322304857
// MI455X (gfx1250) — matrix operand order repaired and checked by reading, not compiled, not run on hardware
//
#include <hip/hip_runtime.h>
#include <hip/hip_bf16.h>
#include <math.h>

typedef __attribute__((ext_vector_type(16))) _Float16 v16h;
typedef __attribute__((ext_vector_type(8)))  _Float16 v8h;
typedef __attribute__((ext_vector_type(8)))  float    v8f;
typedef unsigned int u32x4 __attribute__((ext_vector_type(4)));
typedef int          i32x4 __attribute__((ext_vector_type(4)));
typedef int          i32x8 __attribute__((ext_vector_type(8)));

#define B_   64
#define L_   2048
#define D_   128
#define BL_  (B_ * L_)

#if defined(__gfx1250__) && __has_builtin(__builtin_amdgcn_tensor_load_to_lds)
#define HAVE_TDM 1
#else
#define HAVE_TDM 0
#endif

// --- compile-only probes: report (via stderr warnings) whether the TDM path
// --- is live in the device pass and which builtin arity branch was taken.
#if defined(__HIP_DEVICE_COMPILE__) && defined(__gfx1250__)
#if HAVE_TDM
#warning "CDNA5-PROBE: tensor_load_to_lds builtin AVAILABLE -> TDM path emitted in gru_scan"
#if defined(__clang_major__) && (__clang_major__ >= 23)
#warning "CDNA5-PROBE: using 6-arg tensor_load_to_lds (clang >= 23)"
#else
#warning "CDNA5-PROBE: using 5-arg tensor_load_to_lds (clang < 23)"
#endif
#else
#warning "CDNA5-PROBE: tensor_load_to_lds builtin NOT available -> scalar fallback in gru_scan"
#endif
#endif

union V16H { v16h v; v8h h[2]; };

// ---- CDNA5 async global->LDS copy (ASYNCcnt path), inline asm -------------
__device__ __forceinline__ void async_copy_b128(unsigned lds_off, const void* gaddr) {
    asm volatile("global_load_async_to_lds_b128 %0, %1, off"
                 :: "v"(lds_off), "v"(gaddr) : "memory");
}
__device__ __forceinline__ void wait_asynccnt0() {
    asm volatile("s_wait_asynccnt 0x0" ::: "memory");
}

// ---------------------------------------------------------------------------
// Gather: tok_h[b,l,d] = (f16) E[code[b,l], d]
// ---------------------------------------------------------------------------
__global__ void gather_tok(const int* __restrict__ code,
                           const float* __restrict__ E,
                           _Float16* __restrict__ tok_h) {
    size_t i = (size_t)blockIdx.x * 256 + threadIdx.x;   // over BL_*128
    size_t bl = i >> 7;
    int d = (int)(i & 127);
    tok_h[i] = (_Float16)E[(size_t)code[bl] * D_ + d];
}

// ---------------------------------------------------------------------------
// Pack f32 weight [Wt][K][N] into WMMA B-fragment layout:
//   dst[blk][lane][j], blk = (w*NT + ntile)*KT + ks,
//   lane -> col = ntile*16 + lane%16, K = ks*32 + (lane/16)*8 + (j%8) + (j/8)*16
// ---------------------------------------------------------------------------
__global__ void pack_weights(const float* __restrict__ src,
                             _Float16* __restrict__ dst, int K, int N) {
    int KT = K / 32, NT = N / 16;
    int blk = blockIdx.x;
    int ks = blk % KT;
    int wn = blk / KT;
    int ntile = wn % NT;
    int w = wn / NT;
    int t = threadIdx.x;          // 512 threads
    int lane = t >> 4, j = t & 15;
    int k = ks * 32 + (lane >> 4) * 8 + (j & 7) + (j >> 3) * 16;
    int n = ntile * 16 + (lane & 15);
    dst[(size_t)blk * 512 + lane * 16 + j] = (_Float16)src[((size_t)w * K + k) * N + n];
}

// ---------------------------------------------------------------------------
// xproj = tok @ Wx + b_gru[0]   (M=BL_, K=128, N=192) -> f16
// grid (8192, 3), block 128 (4 waves), ntile = blockIdx.y*4 + wave
// ---------------------------------------------------------------------------
__global__ void gemm_xproj(const _Float16* __restrict__ A,
                           const _Float16* __restrict__ Wp,
                           const float* __restrict__ bgru,
                           _Float16* __restrict__ Xp) {
    int mtile = blockIdx.x;
    int wave = threadIdx.x >> 5, lane = threadIdx.x & 31;
    int ntile = blockIdx.y * 4 + wave;
    const _Float16* arow = A + (size_t)(mtile * 16 + (lane & 15)) * 128 + ((lane >> 4) * 8);
    v8f acc = {};
#pragma unroll
    for (int ks = 0; ks < 4; ++ks) {
        V16H a;
        a.h[0] = *(const v8h*)(arow + ks * 32);
        a.h[1] = *(const v8h*)(arow + ks * 32 + 16);
        v16h bm = *(const v16h*)(Wp + ((size_t)(ntile * 4 + ks) * 32 + lane) * 16);
        acc = __builtin_amdgcn_wmma_f32_16x16x32_f16(false, a.v, false, bm,
                                                     (short)0, acc, false, false);
    }
    int col = ntile * 16 + (lane & 15);
    float bv = bgru[col];
    int rowbase = mtile * 16 + (lane >> 4) * 8;
#pragma unroll
    for (int r = 0; r < 8; ++r)
        Xp[(size_t)(rowbase + r) * 192 + col] = (_Float16)(acc[r] + bv);
}

// ---------------------------------------------------------------------------
// Conv1d SAME (W=8, pad_lo=3) as 8 accumulated WMMA GEMMs.
// 23-row input slab staged into LDS with CDNA5 async-to-LDS copies.
// grid 8192 (16-row m-tile, never crosses batch), block 128 (4 waves=4 ntiles)
// ---------------------------------------------------------------------------
template <int CIN, bool RELU>
__global__ void conv_wmma(const _Float16* __restrict__ in,     // [BL_, CIN]
                          const _Float16* __restrict__ wp,     // packed [8][4][CIN/32][512]
                          const float* __restrict__ bias,      // [64]
                          _Float16* __restrict__ out_h,        // conv1 (relu, f16)
                          float* __restrict__ out_f) {         // conv2 (raw f32)
    constexpr int KT = CIN / 32;
    constexpr int C8 = CIN / 8;
    __shared__ __align__(16) _Float16 smem[23 * CIN];
    int mtile = blockIdx.x;
    int l0 = (mtile * 16) & (L_ - 1);
    int bstart = mtile * 16 - l0;      // b * L_
    int tid = threadIdx.x;
    int wave = tid >> 5, lane = tid & 31;

    for (int i = tid; i < 23 * C8; i += 128) {
        int r = i / C8, c = (i % C8) * 8;
        int l = l0 - 3 + r;
        _Float16* lp = smem + r * CIN + c;
        if (l >= 0 && l < L_) {
            // async DMA directly into LDS, bypassing VGPRs
            async_copy_b128((unsigned)(unsigned long long)lp,
                            in + (size_t)(bstart + l) * CIN + c);
        } else {
            v8h z = {};
            *(v8h*)lp = z;             // zero-pad boundary rows via ds_store
        }
    }
    wait_asynccnt0();
    __syncthreads();

    int ntile = wave;
    v8f acc = {};
    const _Float16* arow = smem + (lane & 15) * CIN + ((lane >> 4) * 8);
#pragma unroll
    for (int w = 0; w < 8; ++w) {
        const _Float16* wpp = wp + (size_t)((w * 4 + ntile) * KT) * 512;
        const _Float16* ar = arow + w * CIN;
#pragma unroll
        for (int ks = 0; ks < KT; ++ks) {
            V16H a;
            a.h[0] = *(const v8h*)(ar + ks * 32);
            a.h[1] = *(const v8h*)(ar + ks * 32 + 16);
            v16h bm = *(const v16h*)(wpp + (size_t)ks * 512 + lane * 16);
            acc = __builtin_amdgcn_wmma_f32_16x16x32_f16(false, a.v, false, bm,
                                                         (short)0, acc, false, false);
        }
    }
    int col = ntile * 16 + (lane & 15);
    float bv = bias[col];
    int rowbase = mtile * 16 + (lane >> 4) * 8;
#pragma unroll
    for (int r = 0; r < 8; ++r) {
        float v = acc[r] + bv;
        if (RELU) {
            v = fmaxf(v, 0.f);
            out_h[(size_t)(rowbase + r) * 64 + col] = (_Float16)v;
        } else {
            out_f[(size_t)(rowbase + r) * 64 + col] = v;
        }
    }
}

// ---------------------------------------------------------------------------
// GRU scan (Keras reset_after=True, gates z,r,h). One block per batch, 64 thr.
// Wh (64x192 f32 = 48KB) pulled into LDS via the Tensor Data Mover when the
// builtin is available (TENSORcnt path), else scalar fallback.
// ---------------------------------------------------------------------------
__global__ void gru_scan(const _Float16* __restrict__ Xp,   // [B_, L_, 192] (has b[0])
                         const float* __restrict__ Wh,      // [64, 192]
                         const float* __restrict__ bgru,    // [2, 192]
                         float* __restrict__ hT) {           // [B_, 64]
    int b = blockIdx.x;
    int u = threadIdx.x;
    __shared__ float h[64];
    __shared__ __align__(16) float wh[64 * 192];
    h[u] = 0.f;
#if HAVE_TDM
    if (u < 32) {   // wave 0 issues one TDM descriptor for the whole 48KB block
        unsigned long long ga = (unsigned long long)Wh;
        unsigned ldsoff = (unsigned)(unsigned long long)&wh[0];
        u32x4 g0;
        g0[0] = 1u;                                   // count=1 valid descriptor
        g0[1] = ldsoff;                               // lds_addr
        g0[2] = (unsigned)ga;                         // global_addr[31:0]
        g0[3] = (unsigned)((ga >> 32) & 0x1FFFFFFu)   // global_addr[56:32]
                | 0x80000000u;                        // type=2 ("image")
        i32x8 g1;
        g1[0] = 0x20000;                              // data_size=2 -> 4 bytes
        g1[1] = (int)(0x3000u << 16);                 // tensor_dim0 = 12288 (lo16)
        g1[2] = (int)(1u << 16);                      // dim0 hi=0, tensor_dim1 = 1
        g1[3] = (int)(0x3000u << 16);                 // dim1 hi=0, tile_dim0 = 12288
        g1[4] = 1;                                    // tile_dim1 = 1
        g1[5] = 0x3000;                               // tensor_dim0_stride = 12288
        g1[6] = 0;
        g1[7] = 0;
        i32x4 z4 = {0, 0, 0, 0};
#if defined(__clang_major__) && (__clang_major__ >= 23)
        i32x8 z8 = {0, 0, 0, 0, 0, 0, 0, 0};
        __builtin_amdgcn_tensor_load_to_lds(g0, g1, z4, z4, z8, 0);
#else
        __builtin_amdgcn_tensor_load_to_lds(g0, g1, z4, z4, 0);
#endif
        __builtin_amdgcn_s_wait_tensorcnt(0);
    }
#else
    for (int i = u; i < 64 * 192; i += 64) wh[i] = Wh[i];
#endif
    __syncthreads();
    float bz = bgru[192 + u], br = bgru[192 + 64 + u], bh = bgru[192 + 128 + u];
    const _Float16* xp = Xp + (size_t)b * L_ * 192;
    for (int t = 0; t < L_; ++t) {
        float xz = (float)xp[u], xr = (float)xp[64 + u], xh = (float)xp[128 + u];
        float hz = bz, hr = br, hc = bh;
#pragma unroll 8
        for (int k = 0; k < 64; ++k) {
            float hk = h[k];
            hz += hk * wh[k * 192 + u];
            hr += hk * wh[k * 192 + 64 + u];
            hc += hk * wh[k * 192 + 128 + u];
        }
        float z = 1.f / (1.f + __expf(-(xz + hz)));
        float r = 1.f / (1.f + __expf(-(xr + hr)));
        float cand = tanhf(xh + r * hc);
        float hn = z * h[u] + (1.f - z) * cand;
        __syncthreads();
        h[u] = hn;
        __syncthreads();
        xp += 192;
    }
    hT[b * 64 + u] = h[u];
}

// ---------------------------------------------------------------------------
// L_feat = l2norm( L2raw * h_t ), one block per (b,l), 64 threads
// ---------------------------------------------------------------------------
__global__ void norm_mul(const float* __restrict__ L2raw,
                         const float* __restrict__ hT,
                         float* __restrict__ Lfeat) {
    int idx = blockIdx.x;           // b*L_ + l
    int b = idx >> 11;
    int k = threadIdx.x;
    float v = L2raw[(size_t)idx * 64 + k] * hT[b * 64 + k];
    __shared__ float red[64];
    red[k] = v * v;
    __syncthreads();
    for (int s = 32; s > 0; s >>= 1) {
        if (k < s) red[k] += red[k + s];
        __syncthreads();
    }
    Lfeat[(size_t)idx * 64 + k] = v * rsqrtf(red[0] + 1e-12f);
}

// ---------------------------------------------------------------------------
// conv3 (W=8, 64->1) + softmax over L. One block per batch, 256 threads.
// ---------------------------------------------------------------------------
__global__ void conv3_softmax(const float* __restrict__ Lfeat,
                              const float* __restrict__ c3w,  // [8*64]
                              const float* __restrict__ c3b,
                              float* __restrict__ alpha) {
    int b = blockIdx.x, tid = threadIdx.x;
    __shared__ float cw[512];
    __shared__ float red[256];
    for (int i = tid; i < 512; i += 256) cw[i] = c3w[i];
    __syncthreads();
    const float* lf = Lfeat + (size_t)b * L_ * 64;
    float local[8];
    float lmax = -1e30f;
#pragma unroll
    for (int j = 0; j < 8; ++j) {
        int l = j * 256 + tid;
        float s = c3b[0];
        for (int w = 0; w < 8; ++w) {
            int li = l + w - 3;
            if (li >= 0 && li < L_) {
                const float* row = lf + (size_t)li * 64;
                const float* ww = cw + w * 64;
#pragma unroll 8
                for (int k = 0; k < 64; ++k) s += row[k] * ww[k];
            }
        }
        local[j] = s;
        lmax = fmaxf(lmax, s);
    }
    red[tid] = lmax;
    __syncthreads();
    for (int s = 128; s > 0; s >>= 1) {
        if (tid < s) red[tid] = fmaxf(red[tid], red[tid + s]);
        __syncthreads();
    }
    float gmax = red[0];
    __syncthreads();
    float lsum = 0.f;
#pragma unroll
    for (int j = 0; j < 8; ++j) { local[j] = __expf(local[j] - gmax); lsum += local[j]; }
    red[tid] = lsum;
    __syncthreads();
    for (int s = 128; s > 0; s >>= 1) {
        if (tid < s) red[tid] += red[tid + s];
        __syncthreads();
    }
    float inv = 1.f / red[0];
#pragma unroll
    for (int j = 0; j < 8; ++j)
        alpha[(size_t)b * L_ + j * 256 + tid] = local[j] * inv;
}

// ---------------------------------------------------------------------------
// n_hat[b,d] = sum_l alpha[b,l] * tok[b,l,d]. One block per batch, 128 thr.
// ---------------------------------------------------------------------------
__global__ void nhat_kernel(const float* __restrict__ alpha,
                            const _Float16* __restrict__ tok,
                            float* __restrict__ nhat) {
    int b = blockIdx.x, d = threadIdx.x;
    const _Float16* t = tok + (size_t)b * L_ * D_ + d;
    const float* al = alpha + (size_t)b * L_;
    float acc = 0.f;
    for (int l = 0; l < L_; ++l) {
        if ((l & 7) == 0) __builtin_prefetch(t + (size_t)(l + 16) * D_, 0, 0);
        acc += al[l] * (float)t[(size_t)l * D_];
    }
    nhat[b * D_ + d] = acc;
}

// ---------------------------------------------------------------------------
// logits = <tok, n_hat> + bias ; softmax over L -> out. Block per batch, 256 thr.
// ---------------------------------------------------------------------------
__global__ void final_softmax(const _Float16* __restrict__ tok,
                              const float* __restrict__ nhat,
                              const int* __restrict__ code,
                              const float* __restrict__ bias_table,
                              float* __restrict__ out) {
    int b = blockIdx.x, tid = threadIdx.x;
    __shared__ float nh[128];
    __shared__ float red[256];
    if (tid < 128) nh[tid] = nhat[b * D_ + tid];
    __syncthreads();
    float local[8];
    float lmax = -1e30f;
#pragma unroll
    for (int j = 0; j < 8; ++j) {
        int l = j * 256 + tid;
        const _Float16* row = tok + ((size_t)b * L_ + l) * D_;
        __builtin_prefetch(row + D_, 0, 0);
        float s = bias_table[code[(size_t)b * L_ + l]];
#pragma unroll 8
        for (int k = 0; k < D_; ++k) s += (float)row[k] * nh[k];
        local[j] = s;
        lmax = fmaxf(lmax, s);
    }
    red[tid] = lmax;
    __syncthreads();
    for (int s = 128; s > 0; s >>= 1) {
        if (tid < s) red[tid] = fmaxf(red[tid], red[tid + s]);
        __syncthreads();
    }
    float gmax = red[0];
    __syncthreads();
    float lsum = 0.f;
#pragma unroll
    for (int j = 0; j < 8; ++j) { local[j] = __expf(local[j] - gmax); lsum += local[j]; }
    red[tid] = lsum;
    __syncthreads();
    for (int s = 128; s > 0; s >>= 1) {
        if (tid < s) red[tid] += red[tid + s];
        __syncthreads();
    }
    float inv = 1.f / red[0];
#pragma unroll
    for (int j = 0; j < 8; ++j)
        out[(size_t)b * L_ + j * 256 + tid] = local[j] * inv;
}

// ---------------------------------------------------------------------------
extern "C" void kernel_launch(void* const* d_in, const int* in_sizes, int n_in,
                              void* d_out, int out_size, void* d_ws, size_t ws_size,
                              hipStream_t stream) {
    const int*   code       = (const int*)d_in[0];
    const float* E          = (const float*)d_in[1];
    const float* bias_table = (const float*)d_in[2];
    const float* Wx         = (const float*)d_in[3];
    const float* Wh         = (const float*)d_in[4];
    const float* bgru       = (const float*)d_in[5];
    const float* c1w        = (const float*)d_in[6];
    const float* c1b        = (const float*)d_in[7];
    const float* c2w        = (const float*)d_in[8];
    const float* c2b        = (const float*)d_in[9];
    const float* c3w        = (const float*)d_in[10];
    const float* c3b        = (const float*)d_in[11];
    float* out = (float*)d_out;

    char* ws = (char*)d_ws;
    size_t off = 0;
    auto alloc = [&](size_t bytes) -> void* {
        void* p = ws + off;
        off = (off + bytes + 255) & ~(size_t)255;
        return p;
    };
    _Float16* tok_h = (_Float16*)alloc((size_t)BL_ * 128 * 2);   // 33.5 MB
    _Float16* xproj = (_Float16*)alloc((size_t)BL_ * 192 * 2);   // 50.3 MB
    float*    L2raw = (float*)xproj;  // alias: conv2 runs after GRU consumed xproj
    _Float16* L1h   = (_Float16*)alloc((size_t)BL_ * 64 * 2);    // 16.8 MB
    float*    Lfeat = (float*)alloc((size_t)BL_ * 64 * 4);       // 33.6 MB
    float*    alpha = (float*)alloc((size_t)BL_ * 4);
    float*    hT    = (float*)alloc((size_t)B_ * 64 * 4);
    float*    nhat  = (float*)alloc((size_t)B_ * 128 * 4);
    _Float16* Wxp   = (_Float16*)alloc((size_t)48 * 512 * 2);
    _Float16* c1wp  = (_Float16*)alloc((size_t)128 * 512 * 2);
    _Float16* c2wp  = (_Float16*)alloc((size_t)64 * 512 * 2);

    gather_tok<<<(BL_ * 128) / 256, 256, 0, stream>>>(code, E, tok_h);
    pack_weights<<<48, 512, 0, stream>>>(Wx, Wxp, 128, 192);
    pack_weights<<<128, 512, 0, stream>>>(c1w, c1wp, 128, 64);
    pack_weights<<<64, 512, 0, stream>>>(c2w, c2wp, 64, 64);

    gemm_xproj<<<dim3(BL_ / 16, 3), 128, 0, stream>>>(tok_h, Wxp, bgru, xproj);
    gru_scan<<<B_, 64, 0, stream>>>(xproj, Wh, bgru, hT);

    conv_wmma<128, true><<<BL_ / 16, 128, 0, stream>>>(tok_h, c1wp, c1b, L1h, nullptr);
    conv_wmma<64, false><<<BL_ / 16, 128, 0, stream>>>(L1h, c2wp, c2b, nullptr, L2raw);

    norm_mul<<<BL_, 64, 0, stream>>>(L2raw, hT, Lfeat);
    conv3_softmax<<<B_, 256, 0, stream>>>(Lfeat, c3w, c3b, alpha);
    nhat_kernel<<<B_, 128, 0, stream>>>(alpha, tok_h, nhat);
    final_softmax<<<B_, 256, 0, stream>>>(tok_h, nhat, code, bias_table, out);
}
